// GeneratorNet_70248485093817
// MI455X (gfx1250) — compile-verified, matrix operand order checked by reading, not run on hardware
//
#include <hip/hip_runtime.h>
#include <hip/hip_bf16.h>

typedef __attribute__((ext_vector_type(16))) _Float16 v16h;
typedef __attribute__((ext_vector_type(8)))  float    v8f;

#define N_IMG 256
#define DOTS  512
#define HC    16
#define WC    128

// Branch-free tanh: tanh(a) = 1 - 2/(exp2(a*2*log2(e)) + 1).
// Lowers to v_exp_f32 + v_rcp_f32 + fma; saturates correctly for |a| large
// (exp2 -> inf -> rcp -> 0 -> 1; exp2 -> 0 -> 1-2 = -1). No EXEC divergence.
__device__ __forceinline__ float tanh_fast(float a) {
    const float K = 2.8853900817779268f;  // 2*log2(e)
    float e = __builtin_amdgcn_exp2f(a * K);
    return 1.0f - 2.0f * __builtin_amdgcn_rcpf(e + 1.0f);
}

// One block per image n. 256 threads = 8 wave32.
// Stage 1: fill LDS with 512 (xs,ys) dot coordinates.
// Stage 2: each wave covers 16 tiles of 16 pixels; the 512-deep dot reduction
//          runs on the matrix pipe: v_wmma_f32_16x16x32_f16 with B = ones.
__global__ __launch_bounds__(256) void
generator_net_kernel(const float* __restrict__ x,
                     const float* __restrict__ W,
                     float* __restrict__ out) {
    __shared__ float2 pts[DOTS];

    const int n   = blockIdx.x;
    const int tid = threadIdx.x;

    // gfx1250 prefetch path (global_prefetch_b8): warm the (shared) weight vector.
    __builtin_prefetch(W + tid, 0, 0);

    // ---- Stage 1: point set -> LDS -------------------------------------
    // W has 1024 floats; thread tid handles W[4t..4t+3] = dots 2t, 2t+1.
    const float  xv = x[n];
    const float4 w4 = reinterpret_cast<const float4*>(W)[tid];
    float2 p0, p1;
    p0.x = (tanh_fast(xv * w4.x) * 0.5f + 0.5f) * (float)WC;
    p0.y = (tanh_fast(xv * w4.y) * 0.5f + 0.5f) * (float)HC;
    p1.x = (tanh_fast(xv * w4.z) * 0.5f + 0.5f) * (float)WC;
    p1.y = (tanh_fast(xv * w4.w) * 0.5f + 0.5f) * (float)HC;
    pts[2 * tid]     = p0;
    pts[2 * tid + 1] = p1;
    __syncthreads();

    // ---- Stage 2: dense field accumulation via WMMA --------------------
    const int lane = tid & 31;
    const int wave = tid >> 5;
    const int hi   = (lane >> 4) & 1;   // lane half (K sub-block / M sub-block)
    const int col  = lane & 15;         // A-matrix row this lane supplies; also C column

    // B = all-ones (32x16 f16): WMMA computes D[m, *] += sum_k A[m, k].
    v16h bones;
#pragma unroll
    for (int i = 0; i < 16; ++i) bones[i] = (_Float16)1.0f;

    for (int ti = 0; ti < 16; ++ti) {
        const int   tile = wave * 16 + ti;          // 0..127
        const float hf   = (float)(tile >> 3);      // pixel row h (0..15)
        const int   w0   = (tile & 7) * 16;         // tile's first pixel column
        const float wf   = (float)(w0 + col);       // this lane's pixel column

        v8f c = {};                                  // f32 accumulators
#pragma unroll 4
        for (int k0 = 0; k0 < DOTS; k0 += 32) {
            const int kb = k0 + hi * 8;
            v16h a;
            // ISA 16-bit A layout (16x32): e<8  -> K = e   (+8*hi)
            //                              e>=8 -> K = e+8 (+8*hi)
#pragma unroll
            for (int e = 0; e < 8; ++e) {
                const float2 pt = pts[kb + e];
                const float dx = wf - pt.x;
                const float dy = hf - pt.y;
                const float d  = __builtin_amdgcn_sqrtf(dx * dx + dy * dy);
                const float wt = fmaxf(0.0f, __builtin_fmaf(-3.0f, d, 3.0f));
                a[e] = (_Float16)wt;
            }
#pragma unroll
            for (int e = 8; e < 16; ++e) {
                const float2 pt = pts[kb + e + 8];
                const float dx = wf - pt.x;
                const float dy = hf - pt.y;
                const float d  = __builtin_amdgcn_sqrtf(dx * dx + dy * dy);
                const float wt = fmaxf(0.0f, __builtin_fmaf(-3.0f, d, 3.0f));
                a[e] = (_Float16)wt;
            }
            // D = A x ones + C : reduces 32 dots for 16 pixels on the matrix pipe.
            c = __builtin_amdgcn_wmma_f32_16x16x32_f16(
                    false, a, false, bones, (short)0, c, false, false);
        }

        // c[r] holds pixel M = r + 8*hi (all 16 C columns identical).
        float v[8];
#pragma unroll
        for (int r = 0; r < 8; ++r) v[r] = tanh_fast(c[r]) * 2.0f - 1.0f;

        if (col == 0) {   // lanes 0 and 16 write the tile's two 8-pixel halves
            float* base = out + n * (HC * WC) + (tile >> 3) * WC + w0 + hi * 8;
            reinterpret_cast<float4*>(base)[0] = make_float4(v[0], v[1], v[2], v[3]);
            reinterpret_cast<float4*>(base)[1] = make_float4(v[4], v[5], v[6], v[7]);
        }
    }
}

extern "C" void kernel_launch(void* const* d_in, const int* in_sizes, int n_in,
                              void* d_out, int out_size, void* d_ws, size_t ws_size,
                              hipStream_t stream) {
    (void)in_sizes; (void)n_in; (void)out_size; (void)d_ws; (void)ws_size;
    const float* x = (const float*)d_in[0];   // [256]
    const float* W = (const float*)d_in[1];   // [1024]
    float* out = (float*)d_out;               // [256*16*128]
    generator_net_kernel<<<dim3(N_IMG), dim3(256), 0, stream>>>(x, W, out);
}
